// TransNeXt_24799141167827
// MI455X (gfx1250) — compile-verified, matrix operand order checked by reading, not run on hardware
//
#include <hip/hip_runtime.h>
#include <hip/hip_bf16.h>
#include <math.h>

typedef __attribute__((ext_vector_type(16))) _Float16 v16h;
typedef __attribute__((ext_vector_type(8)))  float    v8f;

#define B_    8
#define N_    1024
#define DIM_  768
#define NH_   12
#define HD_   64
#define HID_  2048
#define ROWS_ (B_ * N_)        // 8192
#define NQKV_ (3 * DIM_)       // 2304
#define NFC1_ (2 * HID_)       // 4096

// ---------------------------------------------------------------------------
// Fragment loader.
// A-matrix 16x32 f16 (contiguous k order used by this code): lane m in [0,16) holds row (row0+m),
// K = k0..k0+15; lane m+16 holds the same row, K = k0+16..k0+31.
// B-matrix 32x16 f16: lane n holds column n; halves = contraction rows
// 0..15 (lanes<16) / 16..31 (lanes>=16).  For C = X * W^T (W row-major
// [Ncols x K]) and for Q*K^T / P*V with transposed V, both A and B
// fragments are contiguous 32-byte row loads with this one helper.
// ---------------------------------------------------------------------------
__device__ __forceinline__ v16h ldfrag(const _Float16* __restrict__ base,
                                       int row0, int k0, int ld) {
  const int lane = threadIdx.x & 31;
  const int r    = lane & 15;
  const int seg  = (lane >> 4) << 4;
  const _Float16* p = base + (size_t)(row0 + r) * (size_t)ld + (size_t)(k0 + seg);
  return *(const v16h*)p;
}

// ---------------------------------------------------------------------------
// Generic f32 -> f16 conversion (weights / activations).
// ---------------------------------------------------------------------------
__global__ __launch_bounds__(256) void cvt_f16_kernel(const float* __restrict__ src,
                                                      _Float16* __restrict__ dst, int n) {
  int i = blockIdx.x * 256 + threadIdx.x;
  if (i < n) dst[i] = (_Float16)src[i];
}

// ---------------------------------------------------------------------------
// LayerNorm over DIM_=768, one block (256 thr = 8 waves) per row, f16 output.
// ---------------------------------------------------------------------------
__global__ __launch_bounds__(256) void ln_kernel(const float* __restrict__ x,
                                                 const float* __restrict__ wln,
                                                 const float* __restrict__ bln,
                                                 _Float16* __restrict__ out) {
  const int row  = blockIdx.x;
  const int tid  = threadIdx.x;
  const int lane = tid & 31;
  const int w    = tid >> 5;
  const float* px = x + (size_t)row * DIM_;

  float s = 0.f, ss = 0.f;
  for (int i = tid; i < DIM_; i += 256) { float v = px[i]; s += v; ss += v * v; }
  for (int m = 16; m >= 1; m >>= 1) { s += __shfl_xor(s, m, 32); ss += __shfl_xor(ss, m, 32); }

  __shared__ float shs[8], shss[8];
  __shared__ float mu_s, rstd_s;
  if (lane == 0) { shs[w] = s; shss[w] = ss; }
  __syncthreads();
  if (tid == 0) {
    float S = 0.f, SS = 0.f;
    for (int i = 0; i < 8; ++i) { S += shs[i]; SS += shss[i]; }
    float mu = S / (float)DIM_;
    float var = SS / (float)DIM_ - mu * mu;
    mu_s = mu;
    rstd_s = rsqrtf(var + 1e-5f);
  }
  __syncthreads();
  const float mu = mu_s, rstd = rstd_s;
  _Float16* po = out + (size_t)row * DIM_;
  for (int i = tid; i < DIM_; i += 256)
    po[i] = (_Float16)((px[i] - mu) * rstd * wln[i] + bln[i]);
}

// ---------------------------------------------------------------------------
// WMMA GEMM: C[M x Nc] = A_f16[M x K] * W_f16[Nc x K]^T  (+ epilogue)
// Block = 256 thr = 8 waves; tile 128x128; wave sub-tile 32x64.
// mode 0: out = acc + bias                       (qkv raw, fp32)
// mode 1: out = acc + bias + add                 (proj+residual / fc2+residual)
// mode 2: fc1 split: col<HID -> abuf[b][col][n], col>=HID -> gbuf[row][col-HID]
// ---------------------------------------------------------------------------
__global__ __launch_bounds__(256) void gemm16_kernel(
    const _Float16* __restrict__ A, const _Float16* __restrict__ W,
    const float* __restrict__ bias, const float* __restrict__ add,
    float* __restrict__ out, float* __restrict__ out2,
    int Nc, int K, int mode) {
  const int tid  = threadIdx.x;
  const int lane = tid & 31;
  const int w    = tid >> 5;
  const int wm   = w & 3, wn = w >> 2;
  const int row0 = blockIdx.y * 128 + wm * 32;
  const int col0 = blockIdx.x * 128 + wn * 64;

  v8f c[2][4];
  for (int i = 0; i < 2; ++i)
    for (int t = 0; t < 4; ++t)
      for (int j = 0; j < 8; ++j) c[i][t][j] = 0.f;

  for (int k0 = 0; k0 < K; k0 += 32) {
    if (k0 + 32 < K)
      __builtin_prefetch(A + (size_t)row0 * K + (size_t)(k0 + 32), 0, 0);
    v16h a0 = ldfrag(A, row0,      k0, K);
    v16h a1 = ldfrag(A, row0 + 16, k0, K);
    v16h b0 = ldfrag(W, col0,      k0, K);
    v16h b1 = ldfrag(W, col0 + 16, k0, K);
    v16h b2 = ldfrag(W, col0 + 32, k0, K);
    v16h b3 = ldfrag(W, col0 + 48, k0, K);
    c[0][0] = __builtin_amdgcn_wmma_f32_16x16x32_f16(false, a0, false, b0, (short)0, c[0][0], false, false);
    c[0][1] = __builtin_amdgcn_wmma_f32_16x16x32_f16(false, a0, false, b1, (short)0, c[0][1], false, false);
    c[0][2] = __builtin_amdgcn_wmma_f32_16x16x32_f16(false, a0, false, b2, (short)0, c[0][2], false, false);
    c[0][3] = __builtin_amdgcn_wmma_f32_16x16x32_f16(false, a0, false, b3, (short)0, c[0][3], false, false);
    c[1][0] = __builtin_amdgcn_wmma_f32_16x16x32_f16(false, a1, false, b0, (short)0, c[1][0], false, false);
    c[1][1] = __builtin_amdgcn_wmma_f32_16x16x32_f16(false, a1, false, b1, (short)0, c[1][1], false, false);
    c[1][2] = __builtin_amdgcn_wmma_f32_16x16x32_f16(false, a1, false, b2, (short)0, c[1][2], false, false);
    c[1][3] = __builtin_amdgcn_wmma_f32_16x16x32_f16(false, a1, false, b3, (short)0, c[1][3], false, false);
  }

  const int half = lane >> 4, cl = lane & 15;
  for (int i = 0; i < 2; ++i) {
    for (int t = 0; t < 4; ++t) {
      #pragma unroll
      for (int j = 0; j < 8; ++j) {
        const int rr = row0 + 16 * i + 8 * half + j;
        const int cc = col0 + 16 * t + cl;
        float v = c[i][t][j] + bias[cc];
        if (mode == 0) {
          out[(size_t)rr * Nc + cc] = v;
        } else if (mode == 2) {
          const int bb = rr >> 10, nn = rr & 1023;
          if (cc < HID_) out[(((size_t)bb * HID_ + cc) << 10) + nn] = v;
          else           out2[(size_t)rr * HID_ + (cc - HID_)] = v;
        } else {
          out[(size_t)rr * Nc + cc] = v + add[(size_t)rr * Nc + cc];
        }
      }
    }
  }
}

// ---------------------------------------------------------------------------
// QKV transform: l2norm(q)+query_emb scaled by softplus(T)*log(N) -> f16;
// l2norm(k) -> f16; v -> f16 transposed [bh][hd][n].  64 thr per (b,h,n).
// ---------------------------------------------------------------------------
__global__ __launch_bounds__(64) void qkv_transform_kernel(
    const float* __restrict__ qkv, const float* __restrict__ temp,
    const float* __restrict__ qemb, _Float16* __restrict__ qh,
    _Float16* __restrict__ kh, _Float16* __restrict__ vt) {
  const int id = blockIdx.x;
  const int n  = id & 1023;
  const int h  = (id >> 10) % NH_;
  const int b  = id / (NH_ << 10);
  const int d  = threadIdx.x;

  const size_t row = (size_t)(b * N_ + n) * NQKV_;
  const float q = qkv[row + h * HD_ + d];
  const float k = qkv[row + DIM_ + h * HD_ + d];
  const float v = qkv[row + 2 * DIM_ + h * HD_ + d];

  float sq = q * q, sk = k * k;
  for (int m = 16; m >= 1; m >>= 1) { sq += __shfl_xor(sq, m, 32); sk += __shfl_xor(sk, m, 32); }
  __shared__ float sh[4];
  if ((threadIdx.x & 31) == 0) { sh[threadIdx.x >> 5] = sq; sh[2 + (threadIdx.x >> 5)] = sk; }
  __syncthreads();
  const float nq = sqrtf(sh[0] + sh[1]);
  const float nk = sqrtf(sh[2] + sh[3]);
  const float scale = log1pf(__expf(temp[h])) * logf((float)N_);
  const float qe = (q / fmaxf(nq, 1e-12f) + qemb[h * HD_ + d]) * scale;
  const float kn = k / fmaxf(nk, 1e-12f);

  const size_t bh = (size_t)(b * NH_ + h);
  qh[(bh * N_ + n) * HD_ + d] = (_Float16)qe;
  kh[(bh * N_ + n) * HD_ + d] = (_Float16)kn;
  vt[(bh * HD_ + d) * N_ + n] = (_Float16)v;
}

// ---------------------------------------------------------------------------
// CPB MLP: tb[row][h] = (relu(table @ W1^T + b1) @ W2^T + b2)[h]; block=512.
// ---------------------------------------------------------------------------
__global__ __launch_bounds__(512) void cpb_kernel(
    const float* __restrict__ table, const float* __restrict__ w1,
    const float* __restrict__ b1, const float* __restrict__ w2,
    const float* __restrict__ b2, float* __restrict__ tb) {
  const int r = blockIdx.x;
  const int j = threadIdx.x;
  __shared__ float hid[512];
  const float t0 = table[r * 2 + 0], t1 = table[r * 2 + 1];
  hid[j] = fmaxf(t0 * w1[j * 2 + 0] + t1 * w1[j * 2 + 1] + b1[j], 0.f);
  __syncthreads();
  if (j < NH_) {
    float acc = b2[j];
    for (int i = 0; i < 512; ++i) acc += hid[i] * w2[j * 512 + i];
    tb[r * NH_ + j] = acc;
  }
}

// ---------------------------------------------------------------------------
// Flash attention: grid (B*NH, N/128); 8 waves; each wave owns 16 queries.
// Scores via WMMA (Q f16 x K^T f16), online softmax with 16-lane shfl
// reductions, P staged through per-wave LDS (C-layout -> A-layout), then
// O += P * V via WMMA with transposed V.  Bias gathered from tb via rpi.
// ---------------------------------------------------------------------------
__global__ __launch_bounds__(256) void attn_kernel(
    const _Float16* __restrict__ Qh, const _Float16* __restrict__ Kh,
    const _Float16* __restrict__ Vt, const float* __restrict__ tb,
    const int* __restrict__ rpi, _Float16* __restrict__ out16) {
  const int bh = blockIdx.x;
  const int b = bh / NH_, h = bh % NH_;
  const int tid = threadIdx.x, lane = tid & 31, w = tid >> 5;
  const int q0 = blockIdx.y * 128 + w * 16;
  const int half = lane >> 4, cl = lane & 15;

  const _Float16* Q  = Qh + (size_t)bh * N_ * HD_;
  const _Float16* Kp = Kh + (size_t)bh * N_ * HD_;
  const _Float16* Vp = Vt + (size_t)bh * HD_ * N_;

  __shared__ _Float16 lds[8][16 * 32];

  const v16h qa0 = ldfrag(Q, q0, 0, HD_);
  const v16h qa1 = ldfrag(Q, q0, 32, HD_);

  v8f o[4];
  float m[8], l[8];
  for (int t = 0; t < 4; ++t)
    for (int j = 0; j < 8; ++j) o[t][j] = 0.f;
  for (int j = 0; j < 8; ++j) { m[j] = -1e30f; l[j] = 0.f; }

  for (int kc = 0; kc < N_; kc += 32) {
    v8f s0, s1;
    for (int j = 0; j < 8; ++j) { s0[j] = 0.f; s1[j] = 0.f; }
    v16h kb00 = ldfrag(Kp, kc,      0,  HD_);
    v16h kb01 = ldfrag(Kp, kc,      32, HD_);
    v16h kb10 = ldfrag(Kp, kc + 16, 0,  HD_);
    v16h kb11 = ldfrag(Kp, kc + 16, 32, HD_);
    s0 = __builtin_amdgcn_wmma_f32_16x16x32_f16(false, qa0, false, kb00, (short)0, s0, false, false);
    s0 = __builtin_amdgcn_wmma_f32_16x16x32_f16(false, qa1, false, kb01, (short)0, s0, false, false);
    s1 = __builtin_amdgcn_wmma_f32_16x16x32_f16(false, qa0, false, kb10, (short)0, s1, false, false);
    s1 = __builtin_amdgcn_wmma_f32_16x16x32_f16(false, qa1, false, kb11, (short)0, s1, false, false);

    #pragma unroll
    for (int j = 0; j < 8; ++j) {
      const int gq = q0 + 8 * half + j;
      const float b0v = tb[rpi[(size_t)gq * N_ + kc + cl] * NH_ + h];
      const float b1v = tb[rpi[(size_t)gq * N_ + kc + 16 + cl] * NH_ + h];
      const float sv0 = s0[j] + b0v;
      const float sv1 = s1[j] + b1v;
      float mx = fmaxf(sv0, sv1);
      for (int msk = 1; msk < 16; msk <<= 1) mx = fmaxf(mx, __shfl_xor(mx, msk, 32));
      const float mn = fmaxf(m[j], mx);
      const float alpha = __expf(m[j] - mn);
      const float e0 = __expf(sv0 - mn);
      const float e1 = __expf(sv1 - mn);
      float rs = e0 + e1;
      for (int msk = 1; msk < 16; msk <<= 1) rs += __shfl_xor(rs, msk, 32);
      m[j] = mn;
      l[j] = l[j] * alpha + rs;
      for (int t = 0; t < 4; ++t) o[t][j] *= alpha;
      const int r = 8 * half + j;
      lds[w][r * 32 + cl]      = (_Float16)e0;
      lds[w][r * 32 + 16 + cl] = (_Float16)e1;
    }
    asm volatile("s_wait_dscnt 0" ::: "memory");
    __builtin_amdgcn_wave_barrier();
    const v16h pa = *(const v16h*)&lds[w][cl * 32 + half * 16];
    #pragma unroll
    for (int t = 0; t < 4; ++t) {
      v16h vb = ldfrag(Vp, t * 16, kc, N_);
      o[t] = __builtin_amdgcn_wmma_f32_16x16x32_f16(false, pa, false, vb, (short)0, o[t], false, false);
    }
    __builtin_amdgcn_wave_barrier();
  }

  for (int t = 0; t < 4; ++t) {
    #pragma unroll
    for (int j = 0; j < 8; ++j) {
      const int gq = q0 + 8 * half + j;
      const float val = o[t][j] / fmaxf(l[j], 1e-20f);
      out16[((size_t)(b * N_ + gq)) * DIM_ + h * HD_ + t * 16 + cl] = (_Float16)val;
    }
  }
}

// ---------------------------------------------------------------------------
// Depthwise 3x3 conv (SAME) + exact GELU, gated by g; f16 output for fc2.
// abuf layout [b][c][y*32+x]; g layout [row][c].
// ---------------------------------------------------------------------------
__global__ __launch_bounds__(256) void dwgelu_kernel(
    const float* __restrict__ abuf, const float* __restrict__ dww,
    const float* __restrict__ dwb, const float* __restrict__ gbuf,
    _Float16* __restrict__ h16) {
  const int idx = blockIdx.x * 256 + threadIdx.x;
  if (idx >= B_ * HID_ * N_) return;
  const int n = idx & 1023;
  const int c = (idx >> 10) & (HID_ - 1);
  const int b = idx >> 21;
  const int y = n >> 5, x = n & 31;

  const float* ap = abuf + (((size_t)(b * HID_ + c)) << 10);
  float s = dwb[c];
  #pragma unroll
  for (int dy = -1; dy <= 1; ++dy) {
    #pragma unroll
    for (int dx = -1; dx <= 1; ++dx) {
      const int yy = y + dy, xx = x + dx;
      if (yy >= 0 && yy < 32 && xx >= 0 && xx < 32)
        s += ap[yy * 32 + xx] * dww[c * 9 + (dy + 1) * 3 + (dx + 1)];
    }
  }
  const float ge = 0.5f * s * (1.f + erff(s * 0.70710678118f));
  const float g = gbuf[((size_t)(b * N_ + n)) * HID_ + c];
  h16[((size_t)(b * N_ + n)) * HID_ + c] = (_Float16)(ge * g);
}

// ---------------------------------------------------------------------------
// Host launcher
// ---------------------------------------------------------------------------
extern "C" void kernel_launch(void* const* d_in, const int* in_sizes, int n_in,
                              void* d_out, int out_size, void* d_ws, size_t ws_size,
                              hipStream_t stream) {
  const float* x      = (const float*)d_in[0];
  const int*   rpi    = (const int*)d_in[1];
  const float* table  = (const float*)d_in[2];
  const float* qkv_w  = (const float*)d_in[5];
  const float* qkv_b  = (const float*)d_in[6];
  const float* temp   = (const float*)d_in[7];
  const float* qemb   = (const float*)d_in[8];
  const float* c1w    = (const float*)d_in[9];
  const float* c1b    = (const float*)d_in[10];
  const float* c2w    = (const float*)d_in[11];
  const float* c2b    = (const float*)d_in[12];
  const float* proj_w = (const float*)d_in[13];
  const float* proj_b = (const float*)d_in[14];
  const float* n1w    = (const float*)d_in[15];
  const float* n1b    = (const float*)d_in[16];
  const float* n2w    = (const float*)d_in[17];
  const float* n2b    = (const float*)d_in[18];
  const float* fc1w   = (const float*)d_in[19];
  const float* fc1b   = (const float*)d_in[20];
  const float* dww    = (const float*)d_in[21];
  const float* dwb    = (const float*)d_in[22];
  const float* fc2w   = (const float*)d_in[23];
  const float* fc2b   = (const float*)d_in[24];

  const int table_rows = in_sizes[2] / 2;

  char* ws = (char*)d_ws;
  size_t off = 0;
  auto alloc = [&](size_t bytes) -> char* {
    char* p = ws + off;
    off = (off + bytes + 255) & ~(size_t)255;
    return p;
  };

  _Float16* wq16  = (_Float16*)alloc((size_t)NQKV_ * DIM_ * 2);
  _Float16* wp16  = (_Float16*)alloc((size_t)DIM_ * DIM_ * 2);
  _Float16* wf116 = (_Float16*)alloc((size_t)NFC1_ * DIM_ * 2);
  _Float16* wf216 = (_Float16*)alloc((size_t)DIM_ * HID_ * 2);
  _Float16* xn16  = (_Float16*)alloc((size_t)ROWS_ * DIM_ * 2);
  float*    qkvr  = (float*)   alloc((size_t)ROWS_ * NQKV_ * 4);
  _Float16* qh16  = (_Float16*)alloc((size_t)B_ * NH_ * N_ * HD_ * 2);
  _Float16* kh16  = (_Float16*)alloc((size_t)B_ * NH_ * N_ * HD_ * 2);
  _Float16* vt16  = (_Float16*)alloc((size_t)B_ * NH_ * N_ * HD_ * 2);
  float*    tbb   = (float*)   alloc((size_t)table_rows * NH_ * 4);
  _Float16* ao16  = (_Float16*)alloc((size_t)ROWS_ * DIM_ * 2);
  float*    x2    = (float*)   alloc((size_t)ROWS_ * DIM_ * 4);
  _Float16* xn216 = (_Float16*)alloc((size_t)ROWS_ * DIM_ * 2);
  float*    abuf  = (float*)   alloc((size_t)B_ * HID_ * N_ * 4);
  float*    gbuf  = (float*)   alloc((size_t)ROWS_ * HID_ * 4);
  _Float16* h16   = (_Float16*)alloc((size_t)ROWS_ * HID_ * 2);

  auto cvt = [&](const float* s, _Float16* d, int n) {
    cvt_f16_kernel<<<(n + 255) / 256, 256, 0, stream>>>(s, d, n);
  };
  cvt(qkv_w, wq16,  NQKV_ * DIM_);
  cvt(proj_w, wp16, DIM_ * DIM_);
  cvt(fc1w,  wf116, NFC1_ * DIM_);
  cvt(fc2w,  wf216, DIM_ * HID_);

  // LN1
  ln_kernel<<<ROWS_, 256, 0, stream>>>(x, n1w, n1b, xn16);

  // QKV GEMM: [8192x768] x [768->2304]
  gemm16_kernel<<<dim3(NQKV_ / 128, ROWS_ / 128), 256, 0, stream>>>(
      xn16, wq16, qkv_b, nullptr, qkvr, nullptr, NQKV_, DIM_, 0);

  // q/k l2norm + embedding + temperature; v transpose
  qkv_transform_kernel<<<B_ * NH_ * N_, 64, 0, stream>>>(qkvr, temp, qemb, qh16, kh16, vt16);

  // CPB bias table
  cpb_kernel<<<table_rows, 512, 0, stream>>>(table, c1w, c1b, c2w, c2b, tbb);

  // Flash attention
  attn_kernel<<<dim3(B_ * NH_, N_ / 128), 256, 0, stream>>>(qh16, kh16, vt16, tbb, rpi, ao16);

  // proj GEMM + residual -> x2
  gemm16_kernel<<<dim3(DIM_ / 128, ROWS_ / 128), 256, 0, stream>>>(
      ao16, wp16, proj_b, x, x2, nullptr, DIM_, DIM_, 1);

  // LN2
  ln_kernel<<<ROWS_, 256, 0, stream>>>(x2, n2w, n2b, xn216);

  // fc1 GEMM with split epilogue (a -> [B,C,H,W], g -> [row,C])
  gemm16_kernel<<<dim3(NFC1_ / 128, ROWS_ / 128), 256, 0, stream>>>(
      xn216, wf116, fc1b, nullptr, abuf, gbuf, NFC1_, DIM_, 2);

  // depthwise 3x3 + GELU * gate -> f16
  dwgelu_kernel<<<(B_ * HID_ * N_ + 255) / 256, 256, 0, stream>>>(abuf, dww, dwb, gbuf, h16);

  // fc2 GEMM + residual -> output (fp32)
  gemm16_kernel<<<dim3(DIM_ / 128, ROWS_ / 128), 256, 0, stream>>>(
      h16, wf216, fc2b, x2, (float*)d_out, nullptr, DIM_, HID_, 1);
}
